// FixedGCN_25967372271653
// MI455X (gfx1250) — matrix operand order repaired and checked by reading, not compiled, not run on hardware
//
#include <hip/hip_runtime.h>
#include <hip/hip_bf16.h>
#include <stdint.h>

// ---------------------------------------------------------------------------
// GCN on MI455X (gfx1250, wave32):
//   layer: hlin = X @ W  (bf16 WMMA, fp32 acc; B staged in LDS via async copy)
//          agg[dst] += hlin[src] * dinv[src]*dinv[dst]   (f32 atomics, L2-resident)
//          h = relu(agg + hlin * dinv^2 + b)             (self-loop folded in)
//   out = h2 @ Wout + bout
// ---------------------------------------------------------------------------

typedef __attribute__((ext_vector_type(16))) __bf16       v16bf;
typedef __attribute__((ext_vector_type(8)))  float        v8f;
typedef __attribute__((ext_vector_type(4)))  unsigned int u32x4;
typedef __attribute__((ext_vector_type(4)))  int          v4i;

union FragAB { v16bf v; u32x4 q[2]; };

__device__ __forceinline__ __bf16 f2bf(float x) {
  union { float f; unsigned u; } c; c.f = x;
  unsigned r = c.u + 0x7fffu + ((c.u >> 16) & 1u);   // round-to-nearest-even
  union { unsigned short s; __bf16 b; } o;
  o.s = (unsigned short)(r >> 16);
  return o.b;
}

// ---- gfx1250 async global->LDS copy (16B per lane), guarded --------------
// Builtin signature (from clang diag): (v4i AS1*, v4i AS3*, imm int, imm int)
#if __has_builtin(__builtin_amdgcn_global_load_async_to_lds_b128)
#define HAS_ASYNC_LDS 1
#endif

__device__ __forceinline__ void lds_fill_b128(const __bf16* gsrc, __bf16* ldst) {
#ifdef HAS_ASYNC_LDS
  __builtin_amdgcn_global_load_async_to_lds_b128(
      (__attribute__((address_space(1))) v4i*)(unsigned long long)(const void*)gsrc,
      (__attribute__((address_space(3))) v4i*)(unsigned)(unsigned long long)(void*)ldst,
      0, 0);
#else
  *(u32x4*)ldst = *(const u32x4*)gsrc;               // load + ds_store_b128 fallback
#endif
}

__device__ __forceinline__ void wait_async_fills() {
#ifdef HAS_ASYNC_LDS
#if __has_builtin(__builtin_amdgcn_s_wait_asynccnt)
  __builtin_amdgcn_s_wait_asynccnt(0);
#else
  asm volatile("s_wait_asynccnt 0x0" ::: "memory");
#endif
#endif
}

// ---- elementwise fp32 -> bf16 ---------------------------------------------
__global__ void k_cvt_bf16(const float* __restrict__ in, __bf16* __restrict__ out, long n) {
  long i = (long)blockIdx.x * blockDim.x + threadIdx.x;
  if (i < n) out[i] = f2bf(in[i]);
}

// ---- W (KxN row-major fp32) -> WT (NxK row-major bf16) --------------------
__global__ void k_cvt_t_bf16(const float* __restrict__ in, __bf16* __restrict__ out, int K, int N) {
  long i = (long)blockIdx.x * blockDim.x + threadIdx.x;
  if (i >= (long)K * N) return;
  int k = (int)(i / N);
  int n = (int)(i - (long)k * N);
  out[(size_t)n * K + k] = f2bf(in[i]);
}

__global__ void k_fill(float* __restrict__ p, float v, int n) {
  int i = blockIdx.x * blockDim.x + threadIdx.x;
  if (i < n) p[i] = v;
}

// deg[dst] += 1 per edge (deg pre-filled with 1.0 for the self-loop)
__global__ void k_deg(const int* __restrict__ dst, float* __restrict__ deg, int E) {
  int e = blockIdx.x * blockDim.x + threadIdx.x;
  if (e < E) unsafeAtomicAdd(deg + dst[e], 1.0f);
}

__global__ void k_dinv(const float* __restrict__ deg, float* __restrict__ dinv, int n) {
  int i = blockIdx.x * blockDim.x + threadIdx.x;
  if (i < n) {
    float d = deg[i];
    dinv[i] = d > 0.0f ? rsqrtf(d) : 0.0f;
  }
}

// ---- GEMM: C[MxN] = A[MxK](bf16) * BT[NxK](bf16)^T (+ bias) ---------------
// Block (256 thr = 8 waves) owns a 128x64 tile: wave w -> rows w*16..w*16+15.
// Per k-step the block stages the shared 64x32 B slice into LDS (one async
// b128 per thread, double-buffered, rows padded to 40 elems -> conflict-free
// ds_load_b128). A fragments come from global, software-pipelined one k-step
// ahead. 4x v_wmma_f32_16x16x32_bf16 per wave per k-step.
// Waves past the last row tile are CLAMPED to the last valid tile (redundant
// compute, store suppressed) so the hot loop has no exec-mask gating and no
// accumulator phi copies; EXEC is trivially all-1s for every WMMA.
#define BROW 40                                      // padded LDS row (elements)
__global__ void k_gemm_bf16(const __bf16* __restrict__ A, const __bf16* __restrict__ BT,
                            const float* __restrict__ bias, float* __restrict__ C,
                            int M, int N, int K) {
  __shared__ __bf16 sB[2][64 * BROW];                // 2 x 5 KB

  const int tid  = threadIdx.x;
  const int lane = tid & 31;
  const int wv   = tid >> 5;                         // 0..7
  const int half = lane >> 4;
  const int l15  = lane & 15;

  const int mtRaw   = blockIdx.x * 8 + wv;           // 16-row tile index
  const int mtLast  = (M >> 4) - 1;                  // M is a multiple of 16
  const bool storeOk = (mtRaw <= mtLast);            // wave-uniform
  const int mt      = storeOk ? mtRaw : mtLast;      // clamp: keep wave active
  const int n0      = blockIdx.y * 64;               // 64-col strip

  // cooperative B fill: thread tid -> row n = tid/4, 8-elem segment seg = tid%4
  const int fn  = tid >> 2;
  const int fs  = (tid & 3) * 8;
  const __bf16* gB = BT + (size_t)(n0 + fn) * K + fs;
  __bf16*       lB0 = &sB[0][fn * BROW + fs];
  __bf16*       lB1 = &sB[1][fn * BROW + fs];

  // A fragment base (ISA 7.12.2 16-bit A layout: lane<16 K{0..7,16..23})
  const __bf16* arow = A + (size_t)(mt * 16 + l15) * K + half * 8;

  v8f acc[4] = {v8f{}, v8f{}, v8f{}, v8f{}};

  // prologue: stage k-slice 0, preload A frag for k=0
  lds_fill_b128(gB, lB0);
  FragAB a;
  a.q[0] = *(const u32x4*)(arow + 0);
  a.q[1] = *(const u32x4*)(arow + 16);
  wait_async_fills();
  __syncthreads();

  int buf = 0;
  for (int kk = 0; kk < K; kk += 32) {
    const bool more = (kk + 32 < K);
    // issue next k-slice fill into the other buffer (overlaps with WMMAs)
    if (more) lds_fill_b128(gB + kk + 32, buf ? lB0 : lB1);
    // prefetch next A fragment (issued early; its wait lands at the barrier)
    FragAB an;
    if (more) {
      an.q[0] = *(const u32x4*)(arow + kk + 32);
      an.q[1] = *(const u32x4*)(arow + kk + 48);
    }
    {
      const __bf16* bbase = &sB[buf][half * 8];      // B layout: lane<16 K{0..7,16..23}
#pragma unroll
      for (int t = 0; t < 4; ++t) {
        FragAB b;
        const __bf16* bp = bbase + (t * 16 + l15) * BROW;
        b.q[0] = *(const u32x4*)(bp);
        b.q[1] = *(const u32x4*)(bp + 16);
        acc[t] = __builtin_amdgcn_wmma_f32_16x16x32_bf16(
            false, a.v, false, b.v, (short)0, acc[t], false, false);
      }
    }
    wait_async_fills();
    __syncthreads();
    a = an;
    buf ^= 1;
  }

  if (!storeOk) return;                              // wave-uniform, one-time
  // C/D layout: VGPR i -> M = i (lanes 0..15) / 8+i (lanes 16..31), N = lane&15
  const int row0 = mt * 16 + half * 8;
#pragma unroll
  for (int t = 0; t < 4; ++t) {
    const int n  = n0 + t * 16 + l15;
    const float bv = bias ? bias[n] : 0.0f;
#pragma unroll
    for (int i = 0; i < 8; ++i)
      C[(size_t)(row0 + i) * N + n] = acc[t][i] + bv;
  }
}

// ---- edge aggregation: one wave per edge ----------------------------------
__global__ void k_agg(const float* __restrict__ h, const int* __restrict__ src,
                      const int* __restrict__ dst, const float* __restrict__ dinv,
                      float* __restrict__ agg, int E, int d) {
  const int wave = blockIdx.x * (blockDim.x >> 5) + (threadIdx.x >> 5);
  const int lane = threadIdx.x & 31;
  if (wave >= E) return;
  const int s = src[wave];
  const int t = dst[wave];
  const float nrm = dinv[s] * dinv[t];
  const float* hs = h   + (size_t)s * d;
  float*       at = agg + (size_t)t * d;
  for (int f = lane; f < d; f += 32)
    unsafeAtomicAdd(at + f, hs[f] * nrm);            // global_atomic_add_f32
}

// ---- epilogue: h = relu(agg + hlin*dinv^2 + b) -> bf16 (feeds next GEMM) --
__global__ void k_post(const float* __restrict__ agg, const float* __restrict__ hlin,
                       const float* __restrict__ dinv, const float* __restrict__ bias,
                       __bf16* __restrict__ out, long total, int dShift, int dMask) {
  long gid = (long)blockIdx.x * blockDim.x + threadIdx.x;
  if (gid >= total) return;
  const int node = (int)(gid >> dShift);
  const int f    = (int)(gid & dMask);
  const float di = dinv[node];
  const float v  = agg[gid] + hlin[gid] * di * di + bias[f];
  out[gid] = f2bf(fmaxf(v, 0.0f));
}

// ---------------------------------------------------------------------------
extern "C" void kernel_launch(void* const* d_in, const int* in_sizes, int n_in,
                              void* d_out, int out_size, void* d_ws, size_t ws_size,
                              hipStream_t stream) {
  const float* x    = (const float*)d_in[0];
  const int*   eidx = (const int*)d_in[1];   // (2,E): src = eidx[0..E), dst = eidx[E..2E)
  const float* W1   = (const float*)d_in[2];
  const float* b1   = (const float*)d_in[3];
  const float* W2   = (const float*)d_in[4];
  const float* b2   = (const float*)d_in[5];
  const float* Wout = (const float*)d_in[6];
  const float* bout = (const float*)d_in[7];
  float*       out  = (float*)d_out;

  const int K    = 512;                       // in_dim = h1 = h2 = 512
  const int M    = in_sizes[0] / K;           // 50000 nodes (multiple of 16)
  const int E    = in_sizes[1] / 2;           // 800000 edges
  const int H    = 512;
  const int NOUT = in_sizes[6] / K;           // 256

  // --- workspace carve-out (256B aligned) ---
  char* wp = (char*)d_ws;
  auto carve = [&](size_t bytes) -> void* {
    void* p = (void*)wp;
    wp += (bytes + 255) & ~(size_t)255;
    return p;
  };
  __bf16* xb   = (__bf16*)carve((size_t)M * K * 2);     // bf16 input activations
  __bf16* wt1  = (__bf16*)carve((size_t)K * H * 2);     // W1^T bf16 (N-major)
  __bf16* wt2  = (__bf16*)carve((size_t)H * H * 2);
  __bf16* wto  = (__bf16*)carve((size_t)H * NOUT * 2);
  float*  hlin = (float*)carve((size_t)M * H * 4);      // pre-aggregation features
  float*  agg  = (float*)carve((size_t)M * H * 4);      // scatter-add target
  __bf16* hb   = (__bf16*)carve((size_t)M * H * 2);     // post-relu bf16 activations
  float*  deg  = (float*)carve((size_t)M * 4);
  float*  dinv = (float*)carve((size_t)M * 4);

  const int T = 256;                                    // 8 waves / block
  auto cdiv = [](long a, long b) -> int { return (int)((a + b - 1) / b); };

  // --- degrees / normalization (self-loop folded: deg starts at 1) ---
  k_fill<<<cdiv(M, T), T, 0, stream>>>(deg, 1.0f, M);
  k_deg <<<cdiv(E, T), T, 0, stream>>>(eidx + E, deg, E);
  k_dinv<<<cdiv(M, T), T, 0, stream>>>(deg, dinv, M);

  // --- one-time precision conversions ---
  k_cvt_bf16  <<<cdiv((long)M * K, T), T, 0, stream>>>(x, xb, (long)M * K);
  k_cvt_t_bf16<<<cdiv((long)K * H, T), T, 0, stream>>>(W1, wt1, K, H);
  k_cvt_t_bf16<<<cdiv((long)H * H, T), T, 0, stream>>>(W2, wt2, H, H);
  k_cvt_t_bf16<<<cdiv((long)H * NOUT, T), T, 0, stream>>>(Wout, wto, H, NOUT);

  const long totalMH = (long)M * H;
  const int  rowBlks = cdiv(M / 16, 8);                 // 8 wave-tiles per block
  const dim3 gridH(rowBlks, H / 64);
  const dim3 gridO(rowBlks, NOUT / 64);
  const int  aggBlks = cdiv(E, 8);                      // 1 wave / edge

  // --- layer 1 ---
  k_gemm_bf16<<<gridH, T, 0, stream>>>(xb, wt1, nullptr, hlin, M, H, K);
  (void)hipMemsetAsync(agg, 0, (size_t)M * H * sizeof(float), stream);
  k_agg <<<aggBlks, T, 0, stream>>>(hlin, eidx, eidx + E, dinv, agg, E, H);
  k_post<<<cdiv(totalMH, T), T, 0, stream>>>(agg, hlin, dinv, b1, hb, totalMH, 9, 511);

  // --- layer 2 ---
  k_gemm_bf16<<<gridH, T, 0, stream>>>(hb, wt2, nullptr, hlin, M, H, H);
  (void)hipMemsetAsync(agg, 0, (size_t)M * H * sizeof(float), stream);
  k_agg <<<aggBlks, T, 0, stream>>>(hlin, eidx, eidx + E, dinv, agg, E, H);
  k_post<<<cdiv(totalMH, T), T, 0, stream>>>(agg, hlin, dinv, b2, hb, totalMH, 9, 511);

  // --- output projection: out = h2 @ Wout + bout (fp32 out, no aggregation) ---
  k_gemm_bf16<<<gridO, T, 0, stream>>>(hb, wto, bout, out, M, NOUT, H);
}
